// FrameTransformer_10299331576266
// MI455X (gfx1250) — matrix operand order repaired and checked by reading, not compiled, not run on hardware
//
#include <hip/hip_runtime.h>
#include <hip/hip_bf16.h>
#include <math.h>

typedef _Float16 v16h __attribute__((ext_vector_type(16)));
typedef float    v8f  __attribute__((ext_vector_type(8)));

#define EPSN 1e-5f

// ---------------------------------------------------------------------------
// Generic batched WMMA GEMM (CDNA5 wave32, v_wmma_f32_16x16x32_f16).
//   D[batch] = alpha * A[batchA] * op(B[batch]) (+ Cadd[batch]) [sqrelu epilogue]
// REQUIRES: M % 16 == 0, N % (16*NT) == 0, K % 32 == 0, rows 32-byte alignable.
// A: [MxK] row-major (lda). B: [KxN] (ldb) or [NxK] row-major if TRANSB.
// C/D: [MxN] row-major (ldc); if transC, element (m,n) stored at n*ldc+m.
// batchA = aMod ? (batch % aMod) : batch  (weight sharing across batch).
// Fragment layouts used here (A and B share contiguous K order):
//   A 16x32 f16: lanes 0-15 row M=lane, K=0..15; lanes16-31 K=16..31
//   B 32x16 f16: lanes 0-15 col N=lane, K=0..15; lanes16-31 K=16..31
//   C/D 16x16 f32: vgpr r -> M = 8*(lane/16)+r, N = lane%16
// Each wave computes a 16 x (16*NT) C tile; A fragment shared across NT.
// TRANSB is a template parameter so each instantiation has a branch-free,
// straight-line K loop (runtime flag previously caused per-iteration
// s_cbranch + dead address chains + register-pressure blowup).
// ---------------------------------------------------------------------------
template <int NT, int TRANSB>
__global__ void wmma_gemm_kernel(const float* __restrict__ A,
                                 const float* __restrict__ B,
                                 const float* __restrict__ Cadd,
                                 float* __restrict__ D,
                                 int M, int N, int K,
                                 int lda, int ldb, int ldc,
                                 long long sA, long long sB, long long sC,
                                 int aMod, int transC,
                                 int addC, int sqrelu, float alpha)
{
  const int wave  = threadIdx.x >> 5;
  const int lane  = threadIdx.x & 31;
  const int ntile = blockIdx.x * (blockDim.x >> 5) + wave;
  const int mtile = blockIdx.y;
  const int batch = blockIdx.z;
  const int n0 = ntile * 16 * NT, m0 = mtile * 16;
  if (n0 >= N) return;                       // wave-uniform: EXEC stays all-ones

  const int rg  = lane >> 4;
  const int l16 = lane & 15;

  const float* Ab = A + (long long)(aMod ? (batch % aMod) : batch) * sA;
  const float* Bb = B + (long long)batch * sB;
  float*       Db = D + (long long)batch * sC;
  const float* Rb = Cadd ? (Cadd + (long long)batch * sC) : (const float*)0;

  const int m = m0 + l16;
  const float* arow = Ab + (long long)m * lda;

  v8f acc[NT];
#pragma unroll
  for (int t = 0; t < NT; ++t) acc[t] = (v8f){};

  for (int k0 = 0; k0 < K; k0 += 32) {
    // ---- A fragment: two contiguous 8-float runs -> 4x b128 loads ----
    v16h af;
    {
      float ta[16];
      const float4* p0 = (const float4*)(arow + k0 + rg * 16);
      const float4* p1 = (const float4*)(arow + k0 + rg * 16 + 8);
      *(float4*)(ta + 0)  = p0[0];
      *(float4*)(ta + 4)  = p0[1];
      *(float4*)(ta + 8)  = p1[0];
      *(float4*)(ta + 12) = p1[1];
#pragma unroll
      for (int j = 0; j < 16; ++j) af[j] = (_Float16)ta[j];
    }
    // ---- B fragments ----
#pragma unroll
    for (int t = 0; t < NT; ++t) {
      v16h bf;
      const int n = n0 + t * 16 + l16;
      if (TRANSB) {                          // 16 contiguous floats -> 4x b128
        float tb[16];
        const float4* p = (const float4*)(Bb + (long long)n * ldb + k0 + rg * 16);
        *(float4*)(tb + 0)  = p[0];
        *(float4*)(tb + 4)  = p[1];
        *(float4*)(tb + 8)  = p[2];
        *(float4*)(tb + 12) = p[3];
#pragma unroll
        for (int j = 0; j < 16; ++j) bf[j] = (_Float16)tb[j];
      } else {                               // strided column, coalesced across lanes
        const float* bp = Bb + (long long)(k0 + rg * 16) * ldb + n;
#pragma unroll
        for (int j = 0; j < 16; ++j) bf[j] = (_Float16)bp[(long long)j * ldb];
      }
      acc[t] = __builtin_amdgcn_wmma_f32_16x16x32_f16(false, af, false, bf,
                                                      (short)0, acc[t], false, false);
    }
  }

#pragma unroll
  for (int t = 0; t < NT; ++t) {
#pragma unroll
    for (int r = 0; r < 8; ++r) {
      int mm = m0 + rg * 8 + r;
      int nn = n0 + t * 16 + l16;
      long long idx = transC ? ((long long)nn * ldc + mm)
                             : ((long long)mm * ldc + nn);
      float v = alpha * acc[t][r];
      if (addC) v += Rb[idx];
      if (sqrelu) v = (v > 0.f) ? v * v : 0.f;
      Db[idx] = v;
    }
  }
}

// ---------------------------------------------------------------------------
// frame_norm: layer-norm over H per (b,c,w), per-(c,h) affine.
// ---------------------------------------------------------------------------
__global__ void frame_norm_kernel(const float* __restrict__ x,
                                  const float* __restrict__ g,
                                  const float* __restrict__ bta,
                                  float* __restrict__ y,
                                  int B, int C, int H, int W)
{
  long long idx = (long long)blockIdx.x * blockDim.x + threadIdx.x;
  long long total = (long long)B * C * W;
  if (idx >= total) return;
  int w = (int)(idx % W);
  long long bc = idx / W;
  int c = (int)(bc % C);
  const float* col = x + bc * (long long)H * W + w;
  float s = 0.f, s2 = 0.f;
  for (int h = 0; h < H; ++h) { float v = col[(long long)h * W]; s += v; s2 += v * v; }
  float mu  = s / (float)H;
  float var = s2 / (float)H - mu * mu;
  float rs  = rsqrtf(var + EPSN);
  float* out = y + bc * (long long)H * W + w;
  const float* gg = g   + (long long)c * H;
  const float* bb = bta + (long long)c * H;
  for (int h = 0; h < H; ++h)
    out[(long long)h * W] = (col[(long long)h * W] - mu) * rs * gg[h] + bb[h];
}

// ---------------------------------------------------------------------------
// conv3x3 (pad 1, stride (sH,1)), weights for this block's co staged in LDS.
// sqreluIn applies square(relu(.)) to the loaded input; accum adds into out.
// ---------------------------------------------------------------------------
__global__ void conv3x3_kernel(const float* __restrict__ in,
                               const float* __restrict__ wgt,
                               float* __restrict__ out,
                               int B, int Cin, int H, int W, int Cout, int Ho,
                               int strideH, int sqreluIn, int accum)
{
  __shared__ float sw[48 * 9];
  const int bc = blockIdx.z;
  const int b = bc / Cout, co = bc % Cout;
  for (int i = threadIdx.x; i < Cin * 9; i += blockDim.x)
    sw[i] = wgt[(long long)co * Cin * 9 + i];
  __syncthreads();

  const long long total = (long long)Ho * W;
  const float* inb = in + (long long)b * Cin * H * W;
  float* outp = out + ((long long)b * Cout + co) * total;
  for (long long t = (long long)blockIdx.x * blockDim.x + threadIdx.x; t < total;
       t += (long long)gridDim.x * blockDim.x) {
    int w  = (int)(t % W);
    int ho = (int)(t / W);
    int hbase = ho * strideH - 1;
    float acc = 0.f;
    for (int ci = 0; ci < Cin; ++ci) {
      const float* ip = inb + (long long)ci * H * W;
      const float* wp = sw + ci * 9;
      for (int kh = 0; kh < 3; ++kh) {
        int hi = hbase + kh;
        if (hi < 0 || hi >= H) continue;
        const float* row = ip + (long long)hi * W;
        for (int kw = 0; kw < 3; ++kw) {
          int wi = w + kw - 1;
          if (wi < 0 || wi >= W) continue;
          float v = row[wi];
          if (sqreluIn) v = (v > 0.f) ? v * v : 0.f;
          acc += v * wp[kh * 3 + kw];
        }
      }
    }
    if (accum) outp[t] += acc; else outp[t] = acc;
  }
}

__global__ void conv1x1_kernel(const float* __restrict__ in,
                               const float* __restrict__ wgt,
                               float* __restrict__ out,
                               int B, int Cin, int H, int W, int Cout, int Ho,
                               int strideH, int accum)
{
  long long total = (long long)B * Cout * Ho * W;
  for (long long t = (long long)blockIdx.x * blockDim.x + threadIdx.x; t < total;
       t += (long long)gridDim.x * blockDim.x) {
    int w = (int)(t % W);
    long long r = t / W;
    int ho = (int)(r % Ho); r /= Ho;
    int co = (int)(r % Cout);
    int b  = (int)(r / Cout);
    int hi = ho * strideH;
    const float* ip = in + (long long)b * Cin * H * W + (long long)hi * W + w;
    const float* wp = wgt + (long long)co * Cin;
    float acc = 0.f;
    for (int ci = 0; ci < Cin; ++ci) acc += ip[(long long)ci * H * W] * wp[ci];
    if (accum) out[t] += acc; else out[t] = acc;
  }
}

// ---------------------------------------------------------------------------
// softmax over rows of 256 (one wave32 per row, 8 elems/lane, shfl reductions)
// ---------------------------------------------------------------------------
__global__ void softmax256_kernel(const float* __restrict__ S,
                                  float* __restrict__ P, long long rows)
{
  const int wid  = threadIdx.x >> 5;
  const int lane = threadIdx.x & 31;
  long long row = (long long)blockIdx.x * (blockDim.x >> 5) + wid;
  if (row >= rows) return;
  const float* s = S + row * 256;
  float* p = P + row * 256;
  float v[8];
  float mx = -1e30f;
#pragma unroll
  for (int i = 0; i < 8; ++i) { v[i] = s[lane + i * 32]; mx = fmaxf(mx, v[i]); }
  for (int off = 16; off > 0; off >>= 1) mx = fmaxf(mx, __shfl_xor(mx, off, 32));
  float sum = 0.f;
#pragma unroll
  for (int i = 0; i < 8; ++i) { v[i] = __expf(v[i] - mx); sum += v[i]; }
  for (int off = 16; off > 0; off >>= 1) sum += __shfl_xor(sum, off, 32);
  float inv = 1.0f / sum;
#pragma unroll
  for (int i = 0; i < 8; ++i) p[lane + i * 32] = v[i] * inv;
}

// ---------------------------------------------------------------------------
// [BC,F,W] -> [BC*NH, W, 32] head split, K-padded to 32 with zeros
// (+ optional interleaved-pair RoPE). Padding makes attention K a multiple
// of 32 so the WMMA GEMM runs guard-free.
// ---------------------------------------------------------------------------
__global__ void qkv_reshape_kernel(const float* __restrict__ y,
                                   float* __restrict__ q,
                                   int BC, int F, int W, int NH, int applyRope)
{
  int d = F / NH;                        // 16
  long long total = (long long)BC * F * W;
  for (long long t = (long long)blockIdx.x * blockDim.x + threadIdx.x; t < total;
       t += (long long)gridDim.x * blockDim.x) {
    int w = (int)(t % W);
    long long r = t / W;
    int g  = (int)(r % F);
    int bc = (int)(r / F);
    int nh = g / d, dd = g % d;
    float v;
    if (applyRope) {
      int pr = dd >> 1;
      float inv = powf(10000.f, -(float)(2 * pr) / (float)d);
      float fr = (float)w * inv;
      float cs = __cosf(fr), sn = __sinf(fr);
      const float* base = y + ((long long)bc * F + (g - (dd & 1))) * W + w;
      float te = base[0];
      float to = base[(long long)W];
      v = (dd & 1) ? (to * cs + te * sn) : (te * cs - to * sn);
    } else {
      v = y[t];
    }
    long long o = (((long long)bc * NH + nh) * W + w) * 32 + dd;
    q[o] = v;
    q[o + 16] = 0.f;                     // zero K-pad
  }
}

// bilinear x2 upsample along H (align_corners) written into concat channels 0..C
__global__ void up2_concat_kernel(const float* __restrict__ x,
                                  float* __restrict__ dst,
                                  int B, int C, int H, int W, int Ctot)
{
  int Ho = 2 * H;
  long long total = (long long)B * C * Ho * W;
  for (long long t = (long long)blockIdx.x * blockDim.x + threadIdx.x; t < total;
       t += (long long)gridDim.x * blockDim.x) {
    int w = (int)(t % W);
    long long r = t / W;
    int ho = (int)(r % Ho); r /= Ho;
    int c  = (int)(r % C);
    int b  = (int)(r / C);
    float cf = (float)ho * (float)(H - 1) / (float)(Ho - 1);
    int i0 = (int)floorf(cf);
    int i1 = min(i0 + 1, H - 1);
    float f = cf - (float)i0;
    const float* xp = x + ((long long)b * C + c) * H * W + w;
    float v = xp[(long long)i0 * W] * (1.f - f) + xp[(long long)i1 * W] * f;
    dst[(((long long)b * Ctot + c) * Ho + ho) * W + w] = v;
  }
}

__global__ void copy_concat_kernel(const float* __restrict__ src,
                                   float* __restrict__ dst,
                                   int B, int C, int H, int W, int Ctot, int coff)
{
  long long total = (long long)B * C * H * W;
  for (long long t = (long long)blockIdx.x * blockDim.x + threadIdx.x; t < total;
       t += (long long)gridDim.x * blockDim.x) {
    int w = (int)(t % W);
    long long r = t / W;
    int h = (int)(r % H); r /= H;
    int c = (int)(r % C);
    int b = (int)(r / C);
    dst[(((long long)b * Ctot + coff + c) * H + h) * W + w] = src[t];
  }
}

// out[b,i,hw] = sum_c wo[i,c] * d1[b,c,hw]   (C = I = 2)
__global__ void final_mix_kernel(const float* __restrict__ d1,
                                 const float* __restrict__ wo,
                                 float* __restrict__ out, int B, long long HW)
{
  long long total = (long long)B * 2 * HW;
  for (long long t = (long long)blockIdx.x * blockDim.x + threadIdx.x; t < total;
       t += (long long)gridDim.x * blockDim.x) {
    long long hw = t % HW;
    long long r = t / HW;
    int i = (int)(r % 2);
    int b = (int)(r / 2);
    const float* dp = d1 + (long long)b * 2 * HW + hw;
    out[t] = wo[i * 2 + 0] * dp[0] + wo[i * 2 + 1] * dp[HW];
  }
}

// ===========================================================================
// Host orchestration
// ===========================================================================
static inline int cdiv(long long a, long long b) { return (int)((a + b - 1) / b); }

struct EncP { const float *ng, *nb, *c1, *c2, *idn; };
struct LayP { const float *n1g,*n1b,*qc,*qpw,*kc,*kpw,*vc,*vpw,*opw,*odw,
                          *n2g,*n2b,*l1pw,*l1dw,*l2pw,*l2dw; };

extern "C" void kernel_launch(void* const* d_in, const int* in_sizes, int n_in,
                              void* d_out, int out_size, void* d_ws, size_t ws_size,
                              hipStream_t stream) {
  (void)in_sizes; (void)n_in; (void)out_size; (void)ws_size;
  const float* X = (const float*)d_in[0];
  int pi = 1;
  auto nxt = [&]() { return (const float*)d_in[pi++]; };
  auto getEnc = [&]() { EncP e; e.ng=nxt(); e.nb=nxt(); e.c1=nxt(); e.c2=nxt(); e.idn=nxt(); return e; };
  EncP enc1p=getEnc(), enc2p=getEnc(), enc3p=getEnc(), enc4p=getEnc(), enc5p=getEnc();
  LayP L[12];
  for (int i = 0; i < 12; ++i) {
    L[i].n1g=nxt(); L[i].n1b=nxt(); L[i].qc=nxt(); L[i].qpw=nxt();
    L[i].kc=nxt();  L[i].kpw=nxt(); L[i].vc=nxt(); L[i].vpw=nxt();
    L[i].opw=nxt(); L[i].odw=nxt(); L[i].n2g=nxt(); L[i].n2b=nxt();
    L[i].l1pw=nxt(); L[i].l1dw=nxt(); L[i].l2pw=nxt(); L[i].l2dw=nxt();
  }
  EncP dec4p=getEnc(), dec3p=getEnc(), dec2p=getEnc(), dec1p=getEnc();
  const float* OutW = nxt();

  // ---- workspace bump allocator (floats) ----
  float* ws = (float*)d_ws;
  size_t off = 0;
  auto alloc = [&](size_t n) { float* p = ws + off; off += n; return p; };
  const size_t M1 = 1048576;
  float* e1   = alloc(M1);                   // [2,2,1024,256]
  float* e2   = alloc(M1);                   // [2,4,512,256]
  float* e3   = alloc(M1);                   // [2,8,256,256]
  float* e4   = alloc(M1);                   // [2,16,128,256]
  float* xbuf = alloc(M1);                   // [2,32,64,256] transformer state
  float* bufN1 = alloc(3145728);             // concat / norm scratch
  float* bufN2 = alloc(3145728);
  float* bufC  = alloc(2097152);             // conv1 output scratch
  float* t_norm = alloc(M1);
  float* t_conv = alloc(M1);
  float* t_pw   = alloc(2097152);            // up to [2,32,128,256]
  float* Qb = alloc(2097152);                // [256][256][32] K-padded
  float* Kb = alloc(2097152);
  float* Vb = alloc(2097152);
  float* t_a = alloc(M1);
  float* z1  = alloc(2097152);
  float* big0 = alloc(16777216);             // qk / softmax ping-pong (67 MB ea)
  float* big1 = alloc(16777216);
  float* d4b = alloc(M1); float* d3b = alloc(M1);
  float* d2b = alloc(M1); float* d1b = alloc(M1);

  auto gemm = [&](const float* A, const float* B, const float* Cadd, float* D,
                  int M, int N, int K, int lda, int ldb, int ldc,
                  long long sA, long long sB, long long sC,
                  int batch, int aMod, int transB, int transC,
                  int addC, int sqrelu, float alpha) {
    dim3 blk(128);
    if ((N & 31) == 0) {
      dim3 grd(cdiv(cdiv(N, 32), 4), cdiv(M, 16), batch);
      if (transB)
        wmma_gemm_kernel<2, 1><<<grd, blk, 0, stream>>>(A, B, Cadd, D, M, N, K,
            lda, ldb, ldc, sA, sB, sC, aMod, transC, addC, sqrelu, alpha);
      else
        wmma_gemm_kernel<2, 0><<<grd, blk, 0, stream>>>(A, B, Cadd, D, M, N, K,
            lda, ldb, ldc, sA, sB, sC, aMod, transC, addC, sqrelu, alpha);
    } else {
      dim3 grd(cdiv(cdiv(N, 16), 4), cdiv(M, 16), batch);
      if (transB)
        wmma_gemm_kernel<1, 1><<<grd, blk, 0, stream>>>(A, B, Cadd, D, M, N, K,
            lda, ldb, ldc, sA, sB, sC, aMod, transC, addC, sqrelu, alpha);
      else
        wmma_gemm_kernel<1, 0><<<grd, blk, 0, stream>>>(A, B, Cadd, D, M, N, K,
            lda, ldb, ldc, sA, sB, sC, aMod, transC, addC, sqrelu, alpha);
    }
  };
  auto fnorm = [&](const float* x, const float* g, const float* b, float* y,
                   int B, int C, int H, int W) {
    long long tot = (long long)B * C * W;
    frame_norm_kernel<<<cdiv(tot, 256), 256, 0, stream>>>(x, g, b, y, B, C, H, W);
  };
  auto conv3 = [&](const float* in, const float* w, float* out, int B, int Cin,
                   int H, int W, int Cout, int Ho, int sH, int sq, int acc) {
    dim3 grd(cdiv((long long)Ho * W, 256), 1, B * Cout);
    conv3x3_kernel<<<grd, 256, 0, stream>>>(in, w, out, B, Cin, H, W, Cout, Ho, sH, sq, acc);
  };
  auto conv1 = [&](const float* in, const float* w, float* out, int B, int Cin,
                   int H, int W, int Cout, int Ho, int sH, int acc) {
    long long tot = (long long)B * Cout * Ho * W;
    conv1x1_kernel<<<cdiv(tot, 256), 256, 0, stream>>>(in, w, out, B, Cin, H, W, Cout, Ho, sH, acc);
  };

  auto run_enc = [&](const float* xin, EncP p, int B, int Cin, int H, int W,
                     int Cout, int down, float* obuf) {
    int s = down ? 2 : 1, Ho = H / s;
    fnorm(xin, p.ng, p.nb, bufN1, B, Cin, H, W);
    conv3(bufN1, p.c1, bufC, B, Cin, H, W, Cout, H, 1, 0, 0);
    conv1(xin, p.idn, obuf, B, Cin, H, W, Cout, Ho, s, 0);
    conv3(bufC, p.c2, obuf, B, Cout, H, W, Cout, Ho, s, 1, 1);
  };
  auto run_dec = [&](const float* xin, const float* skip, EncP p, int B,
                     int Cx, int Hx, int W, int Cs, float* obuf) {
    int Ho = 2 * Hx, Ctot = Cx + Cs;
    up2_concat_kernel<<<cdiv((long long)B*Cx*Ho*W, 256), 256, 0, stream>>>(xin, bufN1, B, Cx, Hx, W, Ctot);
    copy_concat_kernel<<<cdiv((long long)B*Cs*Ho*W, 256), 256, 0, stream>>>(skip, bufN1, B, Cs, Ho, W, Ctot, Cx);
    fnorm(bufN1, p.ng, p.nb, bufN2, B, Ctot, Ho, W);
    conv3(bufN2, p.c1, bufC, B, Ctot, Ho, W, Cs, Ho, 1, 0, 0);
    conv1(bufN1, p.idn, obuf, B, Ctot, Ho, W, Cs, Ho, 1, 0);
    conv3(bufC, p.c2, obuf, B, Cs, Ho, W, Cs, Ho, 1, 1, 1);
  };

  // ---------------- encoders ----------------
  run_enc(X,  enc1p, 2,  2, 1024, 256,  2, 0, e1);
  run_enc(e1, enc2p, 2,  2, 1024, 256,  4, 1, e2);
  run_enc(e2, enc3p, 2,  4,  512, 256,  8, 1, e3);
  run_enc(e3, enc4p, 2,  8,  256, 256, 16, 1, e4);
  run_enc(e4, enc5p, 2, 16,  128, 256, 32, 1, xbuf);

  // ---------------- transformer ----------------
  const int B = 2, TC = 32, F = 64, Wt = 256, NH = 4, Dh = 16;
  const int BC = B * TC;
  const long long FW = (long long)F * Wt;       // 16384
  const long long QS = (long long)Wt * 32;      // 8192: padded per-(b,c,head) stride
  const float scale = 0.125f;                   // 1/sqrt(64)
  float* Sprev = big0;
  float* Scur  = big1;

  for (int li = 0; li < 12; ++li) {
    const LayP& p = L[li];
    fnorm(xbuf, p.n1g, p.n1b, t_norm, B, TC, F, Wt);

    // Q / K / V: conv3x3 -> positionwise (per-channel [64x64] WMMA GEMM) -> head split (+RoPE)
    const float* pcv[3] = { p.qc, p.kc, p.vc };
    const float* ppw[3] = { p.qpw, p.kpw, p.vpw };
    float* dst[3] = { Qb, Kb, Vb };
    for (int qkv = 0; qkv < 3; ++qkv) {
      conv3(t_norm, pcv[qkv], t_conv, B, TC, F, Wt, TC, F, 1, 0, 0);
      gemm(ppw[qkv], t_conv, nullptr, t_pw, F, Wt, F, F, Wt, Wt,
           (long long)F * F, FW, FW, BC, TC, 0, 0, 0, 0, 1.0f);
      qkv_reshape_kernel<<<4096, 256, 0, stream>>>(t_pw, dst[qkv], BC, F, Wt, NH, qkv < 2);
    }

    // S = scale * Q K^T + pqk   (per (b,c,head): 256x256, K padded 16->32)
    gemm(Qb, Kb, (li > 0) ? Sprev : nullptr, Scur, 256, 256, 32, 32, 32, 256,
         QS, QS, 65536, BC * NH, 0, /*transB*/1, 0, li > 0, 0, scale);

    // softmax rows -> P (reuses just-consumed pqk buffer)
    softmax256_kernel<<<cdiv(65536, 4), 128, 0, stream>>>(Scur, Sprev, 65536);
    float* P = Sprev;

    // A = P·V, stored transposed straight into [b,c,h,w] layout (t_a)
    gemm(P, Vb, nullptr, t_a, 256, Dh, 256, 256, 32, Wt,
         65536, QS, 4096, BC * NH, 0, 0, /*transC*/1, 0, 0, 1.0f);

    // out = odw( opw_pw(A) );  x += out
    gemm(p.opw, t_a, nullptr, t_pw, F, Wt, F, F, Wt, Wt,
         (long long)F * F, FW, FW, BC, TC, 0, 0, 0, 0, 1.0f);
    gemm(p.odw, t_pw, xbuf, xbuf, TC, (int)FW, TC, TC, (int)FW, (int)FW,
         0, TC * FW, TC * FW, B, 1, 0, 0, /*addC*/1, 0, 1.0f);

    // MLP: x += l2dw( l2pw( sqrelu( l1dw( l1pw( norm(x) )))))
    fnorm(xbuf, p.n2g, p.n2b, t_norm, B, TC, F, Wt);
    gemm(p.l1pw, t_norm, nullptr, t_pw, 2 * F, Wt, F, F, Wt, Wt,
         (long long)2 * F * F, FW, 2 * FW, BC, TC, 0, 0, 0, 0, 1.0f);
    gemm(p.l1dw, t_pw, nullptr, z1, TC, (int)(2 * FW), TC, TC, (int)(2 * FW), (int)(2 * FW),
         0, TC * 2 * FW, TC * 2 * FW, B, 1, 0, 0, 0, /*sqrelu*/1, 1.0f);
    gemm(p.l2pw, z1, nullptr, t_pw, F, Wt, 2 * F, 2 * F, Wt, Wt,
         (long long)F * 2 * F, 2 * FW, FW, BC, TC, 0, 0, 0, 0, 1.0f);
    gemm(p.l2dw, t_pw, xbuf, xbuf, TC, (int)FW, TC, TC, (int)FW, (int)FW,
         0, TC * FW, TC * FW, B, 1, 0, 0, /*addC*/1, 0, 1.0f);

    float* tmp = Sprev; Sprev = Scur; Scur = tmp;   // carry qk chain
  }

  // ---------------- decoders ----------------
  run_dec(xbuf, e4, dec4p, 2, 32,  64, 256, 16, d4b);
  run_dec(d4b,  e3, dec3p, 2, 16, 128, 256,  8, d3b);
  run_dec(d3b,  e2, dec2p, 2,  8, 256, 256,  4, d2b);
  run_dec(d2b,  e1, dec1p, 2,  4, 512, 256,  2, d1b);

  final_mix_kernel<<<4096, 256, 0, stream>>>(d1b, OutW, (float*)d_out, 2,
                                             (long long)1024 * 256);
}
